// GatedGCNLayer_v2_7069516169368
// MI455X (gfx1250) — matrix operand order repaired and checked by reading, not compiled, not run on hardware
//
#include <hip/hip_runtime.h>

typedef __attribute__((ext_vector_type(16))) __bf16 v16bf;
typedef __attribute__((ext_vector_type(8)))  float  v8f;

static constexpr int kN   = 50000;
static constexpr int kE   = 800000;
static constexpr int kG   = 100;
static constexpr int kD   = 96;
static constexpr int kH   = 192;
static constexpr int kNPG = 500;
static constexpr int kEPG = 8000;
static constexpr int kSlices = 16;   // stats reduction slices per graph

// ---------------- weight prep ----------------

// Wt[o*din + k] = W[k*dout + o]   (bf16 transpose so B-fragments read contiguous K)
__global__ void k_transpose_bf16(const float* __restrict__ W, __bf16* __restrict__ Wt,
                                 int din, int dout) {
  int i = blockIdx.x * blockDim.x + threadIdx.x;
  if (i >= din * dout) return;
  int o = i / din, k = i % din;
  Wt[i] = (__bf16)W[k * dout + o];
}

__global__ void k_bias_cat(const float* __restrict__ a, const float* __restrict__ b,
                           const float* __restrict__ c, const float* __restrict__ d,
                           float* __restrict__ out) {
  int i = threadIdx.x;
  if (i >= 4 * kD) return;
  const float* p[4] = {a, b, c, d};
  out[i] = p[i / kD][i % kD];
}

// ---------------- GraphNorm stats: deterministic two-stage reduction ----------------
// stage 1: grid (G, kSlices); threads 0..95 own a feature column (coalesced across lanes)
__global__ void k_stats_partial(const float* __restrict__ x, int rows,
                                float* __restrict__ psum, float* __restrict__ psq) {
  int g = blockIdx.x, s = blockIdx.y;
  int d = threadIdx.x;
  if (d >= kD) return;
  int chunk = (rows + kSlices - 1) / kSlices;
  int r0 = s * chunk;
  int r1 = min(rows, r0 + chunk);
  const float* base = x + (size_t)g * rows * kD;
  float sum = 0.f, sq = 0.f;
  for (int r = r0; r < r1; ++r) {
    float v = base[(size_t)r * kD + d];
    sum += v; sq += v * v;
  }
  size_t idx = ((size_t)s * kG + g) * kD + d;
  psum[idx] = sum;
  psq[idx]  = sq;
}

// stage 2: one thread per (g,d); var of (x - mean*msc) = E[x^2] - 2*m2*E[x] + m2^2
__global__ void k_stats_final(int rows, const float* __restrict__ psum,
                              const float* __restrict__ psq, const float* __restrict__ msc,
                              float* __restrict__ mean, float* __restrict__ rstd) {
  int i = blockIdx.x * blockDim.x + threadIdx.x;   // g*kD + d
  if (i >= kG * kD) return;
  int d = i % kD;
  float s = 0.f, sq = 0.f;
#pragma unroll
  for (int t = 0; t < kSlices; ++t) {
    s  += psum[(size_t)t * kG * kD + i];
    sq += psq[(size_t)t * kG * kD + i];
  }
  float m   = s / rows;
  float msq = sq / rows;
  float m2  = m * msc[d];
  float var = msq - 2.f * m2 * m + m2 * m2;
  mean[i] = m;
  rstd[i] = rsqrtf(var + 1e-6f);
}

// ---------------- shared device helpers ----------------

// cooperative GraphNorm of a 16-row fp32 tile into bf16 LDS ([16][96])
__device__ __forceinline__ void normalize_tile_to_lds(
    const float* __restrict__ x, int rowBase, int rowsPerG,
    const float* __restrict__ gnw, const float* __restrict__ gnb,
    const float* __restrict__ msc,
    const float* __restrict__ mean, const float* __restrict__ rstd,
    __bf16* __restrict__ xn) {
  for (int i = threadIdx.x; i < 16 * kD; i += blockDim.x) {
    int r = i / kD, d = i % kD;
    int row = rowBase + r;
    int g = row / rowsPerG;               // tiles may straddle graphs (500 % 16 != 0)
    float sub = x[(size_t)row * kD + d] - mean[g * kD + d] * msc[d];
    xn[i] = (__bf16)(gnw[d] * sub * rstd[g * kD + d] + gnb[d]);
  }
  __syncthreads();
}

// A-fragments for K=96 from an LDS tile laid out [16][ldk]
__device__ __forceinline__ void load_a_frags3(const __bf16* __restrict__ xn, int m, int grp,
                                              int ldk, v16bf a[3]) {
  const __bf16* xrow = xn + m * ldk;
#pragma unroll
  for (int kb = 0; kb < 3; ++kb) {
    int b0 = kb * 32 + grp * 16;
#pragma unroll
    for (int t = 0; t < 8; ++t) { a[kb][t] = xrow[b0 + t]; a[kb][8 + t] = xrow[b0 + 8 + t]; }
  }
}

// ---------------- fused node GEMM: norm(h) in LDS, [A|B|C|D] = hn @ Wcat + bcat --------
// block = 8 waves, 16 rows; each wave does 3 of 24 col-tiles; K=96 => 3 WMMAs/tile
__global__ void k_gemm_node_fused(const float* __restrict__ h,
                                  const float* __restrict__ gnw, const float* __restrict__ gnb,
                                  const float* __restrict__ msc,
                                  const float* __restrict__ mean, const float* __restrict__ rstd,
                                  const __bf16* __restrict__ Wt, const float* __restrict__ bias,
                                  float* __restrict__ out) {
  __shared__ __bf16 xn[16 * kD];
  const int rowBase = blockIdx.x * 16;
  normalize_tile_to_lds(h, rowBase, kNPG, gnw, gnb, msc, mean, rstd, xn);

  const int lane = threadIdx.x & 31;
  const int wave = threadIdx.x >> 5;
  const int m   = lane & 15;
  const int grp = lane >> 4;
  v16bf a[3];
  load_a_frags3(xn, m, grp, kD, a);

  for (int ct = wave; ct < 24; ct += 8) {
    int col = ct * 16 + m;
    const __bf16* wrow = Wt + (size_t)col * kD;
    v8f acc = {};
#pragma unroll
    for (int kb = 0; kb < 3; ++kb) {
      v16bf b;
      int koff = kb * 32 + grp * 16;
#pragma unroll
      for (int t = 0; t < 16; ++t) b[t] = wrow[koff + t];
      acc = __builtin_amdgcn_wmma_f32_16x16x32_bf16(false, a[kb], false, b,
                                                    (short)0, acc, false, false);
    }
    float bv = bias[col];
#pragma unroll
    for (int r = 0; r < 8; ++r) {
      int mm = r + grp * 8;
      out[(size_t)(rowBase + mm) * (4 * kD) + col] = acc[r] + bv;
    }
  }
}

// ---------------- fused edge kernel: norm(e) in LDS + Ee WMMA + gather + sigmoid
//                  + scatter atomics + e residual -------------------------------------
// block = 6 waves, 16 edges; wave owns one 16-col tile; K=96 => 3 WMMAs
__global__ void k_edge_gate_fused(const float* __restrict__ e_in,
                                  const float* __restrict__ gnw, const float* __restrict__ gnb,
                                  const float* __restrict__ msc,
                                  const float* __restrict__ mean, const float* __restrict__ rstd,
                                  const __bf16* __restrict__ Wt, const float* __restrict__ Eb,
                                  const int* __restrict__ src, const int* __restrict__ dst,
                                  const float* __restrict__ node_lin,
                                  float* __restrict__ agg_num, float* __restrict__ agg_den,
                                  float* __restrict__ e_stage, const float* __restrict__ gp) {
  __shared__ __bf16 xn[16 * kD];
  const int rowBase = blockIdx.x * 16;
  normalize_tile_to_lds(e_in, rowBase, kEPG, gnw, gnb, msc, mean, rstd, xn);

  const int lane = threadIdx.x & 31;
  const int wave = threadIdx.x >> 5;
  const int m   = lane & 15;
  const int grp = lane >> 4;
  const float gv = gp[0];
  v16bf a[3];
  load_a_frags3(xn, m, grp, kD, a);

  int col = wave * 16 + m;
  const __bf16* wrow = Wt + (size_t)col * kD;
  v8f acc = {};
#pragma unroll
  for (int kb = 0; kb < 3; ++kb) {
    v16bf b;
    int koff = kb * 32 + grp * 16;
#pragma unroll
    for (int t = 0; t < 16; ++t) b[t] = wrow[koff + t];
    acc = __builtin_amdgcn_wmma_f32_16x16x32_bf16(false, a[kb], false, b,
                                                  (short)0, acc, false, false);
  }
  float eb = Eb[col];
#pragma unroll
  for (int r = 0; r < 8; ++r) {
    int mm = r + grp * 8;
    int edge = rowBase + mm;
    int s = src[edge], t = dst[edge];
    float e2 = node_lin[(size_t)s * 384 + 2 * kD + col]   // Ch[src]
             + node_lin[(size_t)t * 384 + 3 * kD + col]   // Dh[dst]
             + acc[r] + eb;                               // Ee
    float sig = 1.0f / (1.0f + expf(-e2));
    float bh  = node_lin[(size_t)s * 384 + kD + col];     // Bh[src]
    atomicAdd(&agg_num[(size_t)t * kD + col], bh * sig);
    atomicAdd(&agg_den[(size_t)t * kD + col], sig);
    e_stage[(size_t)edge * kD + col] = e_in[(size_t)edge * kD + col] + gv * e2;
  }
}

// ---------------- node update: h = h_in + g*(Ah + num/(den+eps)) ----------------
__global__ void k_node_update(const float* __restrict__ h_in, const float* __restrict__ node_lin,
                              const float* __restrict__ agg_num, const float* __restrict__ agg_den,
                              float* __restrict__ h_out, const float* __restrict__ gp) {
  long i = (long)blockIdx.x * blockDim.x + threadIdx.x;
  if (i >= (long)kN * kD) return;
  long row = i / kD; int d = (int)(i % kD);
  float ah = node_lin[row * (4 * kD) + d];
  float h2 = ah + agg_num[i] / (agg_den[i] + 1e-10f);
  h_out[i] = h_in[i] + gp[0] * h2;
}

// ---------------- fused FFN: norm(io) in LDS, 96 -> 192 relu -> 96, in-place residual --
// block = 6 waves, 16 rows; phase1: 12 mid col-tiles (2/wave); phase2: 6 out tiles (1/wave)
__global__ void k_ffn_fused(float* __restrict__ io, int rowsPerG,
                            const float* __restrict__ gnw, const float* __restrict__ gnb,
                            const float* __restrict__ msc,
                            const float* __restrict__ mean, const float* __restrict__ rstd,
                            const __bf16* __restrict__ Wt1, const float* __restrict__ b1,
                            const __bf16* __restrict__ Wt2, const float* __restrict__ b2,
                            const float* __restrict__ gp) {
  __shared__ __bf16 xn[16 * kD];
  __shared__ __bf16 mid[16 * kH];
  const int rowBase = blockIdx.x * 16;
  normalize_tile_to_lds(io, rowBase, rowsPerG, gnw, gnb, msc, mean, rstd, xn);

  const int lane = threadIdx.x & 31;
  const int wave = threadIdx.x >> 5;
  const int m   = lane & 15;
  const int grp = lane >> 4;
  const float gv = gp[0];
  v16bf a[3];
  load_a_frags3(xn, m, grp, kD, a);

#pragma unroll
  for (int tt = 0; tt < 2; ++tt) {
    int ct = wave + tt * 6;
    int col = ct * 16 + m;
    const __bf16* wrow = Wt1 + (size_t)col * kD;
    v8f acc = {};
#pragma unroll
    for (int kb = 0; kb < 3; ++kb) {
      v16bf b;
      int koff = kb * 32 + grp * 16;
#pragma unroll
      for (int t = 0; t < 16; ++t) b[t] = wrow[koff + t];
      acc = __builtin_amdgcn_wmma_f32_16x16x32_bf16(false, a[kb], false, b,
                                                    (short)0, acc, false, false);
    }
    float bb = b1[col];
#pragma unroll
    for (int r = 0; r < 8; ++r) {
      int mm = r + grp * 8;
      float v = acc[r] + bb;
      mid[mm * kH + col] = (__bf16)(v > 0.f ? v : 0.f);
    }
  }
  __syncthreads();

  int col2 = wave * 16 + m;
  const __bf16* w2row = Wt2 + (size_t)col2 * kH;
  v8f acc2 = {};
#pragma unroll
  for (int kb = 0; kb < 6; ++kb) {
    v16bf a2, b;
    int b0 = kb * 32 + grp * 16;
#pragma unroll
    for (int t = 0; t < 8; ++t) {
      a2[t]     = mid[m * kH + b0 + t];
      a2[8 + t] = mid[m * kH + b0 + 8 + t];
    }
    int koff = kb * 32 + grp * 16;
#pragma unroll
    for (int t = 0; t < 16; ++t) b[t] = w2row[koff + t];
    acc2 = __builtin_amdgcn_wmma_f32_16x16x32_bf16(false, a2, false, b,
                                                   (short)0, acc2, false, false);
  }
  float bb2 = b2[col2];
#pragma unroll
  for (int r = 0; r < 8; ++r) {
    int mm = r + grp * 8;
    io[(size_t)(rowBase + mm) * kD + col2] += gv * (acc2[r] + bb2);
  }
}

// ---------------- host ----------------

extern "C" void kernel_launch(void* const* d_in, const int* in_sizes, int n_in,
                              void* d_out, int out_size, void* d_ws, size_t ws_size,
                              hipStream_t stream) {
  const float* h_in = (const float*)d_in[0];
  const float* e_in = (const float*)d_in[1];
  const int*   src  = (const int*)d_in[2];
  const int*   dst  = (const int*)d_in[3];
  // d_in[4..7]: node_seg/edge_seg/node_cnt/edge_cnt (segments contiguous; counts constant)
  const float* Aw = (const float*)d_in[8];   const float* Ab = (const float*)d_in[9];
  const float* Bw = (const float*)d_in[10];  const float* Bb = (const float*)d_in[11];
  const float* Cw = (const float*)d_in[12];  const float* Cb = (const float*)d_in[13];
  const float* Dw = (const float*)d_in[14];  const float* Db = (const float*)d_in[15];
  const float* Ew = (const float*)d_in[16];  const float* Eb = (const float*)d_in[17];
  const float* f1hw = (const float*)d_in[18]; const float* f1hb = (const float*)d_in[19];
  const float* f2hw = (const float*)d_in[20]; const float* f2hb = (const float*)d_in[21];
  const float* f1ew = (const float*)d_in[22]; const float* f1eb = (const float*)d_in[23];
  const float* f2ew = (const float*)d_in[24]; const float* f2eb = (const float*)d_in[25];
  const float* n1h_gnw = (const float*)d_in[26]; const float* n1h_gnb = (const float*)d_in[27];
  const float* n1h_msc = (const float*)d_in[28];
  const float* n1e_gnw = (const float*)d_in[29]; const float* n1e_gnb = (const float*)d_in[30];
  const float* n1e_msc = (const float*)d_in[31];
  const float* n2h_gnw = (const float*)d_in[32]; const float* n2h_gnb = (const float*)d_in[33];
  const float* n2h_msc = (const float*)d_in[34];
  const float* n2e_gnw = (const float*)d_in[35]; const float* n2e_gnb = (const float*)d_in[36];
  const float* n2e_msc = (const float*)d_in[37];
  const float* gp = (const float*)d_in[38];

  float* h_out = (float*)d_out;                   // [N,96]
  float* e_out = (float*)d_out + (size_t)kN * kD; // [E,96]

  // workspace carve-up
  char* ws = (char*)d_ws;
  size_t off = 0;
  auto take = [&](size_t bytes) -> char* {
    char* p = ws + off;
    off = (off + bytes + 255) & ~(size_t)255;
    return p;
  };
  __bf16* wt_abcd = (__bf16*)take((size_t)4 * kD * kD * 2);
  __bf16* wt_e    = (__bf16*)take((size_t)kD * kD * 2);
  __bf16* wt_f1h  = (__bf16*)take((size_t)kH * kD * 2);
  __bf16* wt_f2h  = (__bf16*)take((size_t)kD * kH * 2);
  __bf16* wt_f1e  = (__bf16*)take((size_t)kH * kD * 2);
  __bf16* wt_f2e  = (__bf16*)take((size_t)kD * kH * 2);
  float*  bias_abcd = (float*)take((size_t)4 * kD * 4);
  float*  psum    = (float*)take((size_t)kSlices * kG * kD * 4);
  float*  psq     = (float*)take((size_t)kSlices * kG * kD * 4);
  float*  mean_n  = (float*)take((size_t)kG * kD * 4);
  float*  rstd_n  = (float*)take((size_t)kG * kD * 4);
  float*  mean_ed = (float*)take((size_t)kG * kD * 4);
  float*  rstd_ed = (float*)take((size_t)kG * kD * 4);
  float*  node_lin = (float*)take((size_t)kN * 4 * kD * 4);
  float*  agg_num  = (float*)take((size_t)kN * kD * 4);
  float*  agg_den  = (float*)take((size_t)kN * kD * 4);

  hipMemsetAsync(agg_num, 0, (size_t)kN * kD * 4, stream);
  hipMemsetAsync(agg_den, 0, (size_t)kN * kD * 4, stream);

  // weight prep (bf16, transposed to [out][k])
  int te = kD * kD;
  k_transpose_bf16<<<(te + 255) / 256, 256, 0, stream>>>(Aw, wt_abcd + 0 * te, kD, kD);
  k_transpose_bf16<<<(te + 255) / 256, 256, 0, stream>>>(Bw, wt_abcd + 1 * te, kD, kD);
  k_transpose_bf16<<<(te + 255) / 256, 256, 0, stream>>>(Cw, wt_abcd + 2 * te, kD, kD);
  k_transpose_bf16<<<(te + 255) / 256, 256, 0, stream>>>(Dw, wt_abcd + 3 * te, kD, kD);
  k_transpose_bf16<<<(te + 255) / 256, 256, 0, stream>>>(Ew, wt_e, kD, kD);
  int tf = kD * kH;
  k_transpose_bf16<<<(tf + 255) / 256, 256, 0, stream>>>(f1hw, wt_f1h, kD, kH);
  k_transpose_bf16<<<(tf + 255) / 256, 256, 0, stream>>>(f2hw, wt_f2h, kH, kD);
  k_transpose_bf16<<<(tf + 255) / 256, 256, 0, stream>>>(f1ew, wt_f1e, kD, kH);
  k_transpose_bf16<<<(tf + 255) / 256, 256, 0, stream>>>(f2ew, wt_f2e, kH, kD);
  k_bias_cat<<<1, 4 * kD, 0, stream>>>(Ab, Bb, Cb, Db, bias_abcd);

  dim3 sgrid(kG, kSlices);
  int statsThreads = 128;
  int finBlocks = (kG * kD + 255) / 256;

  // GraphNorm 1 stats
  k_stats_partial<<<sgrid, statsThreads, 0, stream>>>(h_in, kNPG, psum, psq);
  k_stats_final<<<finBlocks, 256, 0, stream>>>(kNPG, psum, psq, n1h_msc, mean_n, rstd_n);
  k_stats_partial<<<sgrid, statsThreads, 0, stream>>>(e_in, kEPG, psum, psq);
  k_stats_final<<<finBlocks, 256, 0, stream>>>(kEPG, psum, psq, n1e_msc, mean_ed, rstd_ed);

  // node GEMMs (A|B|C|D, norm fused), then fused edge gate + scatter + e residual
  k_gemm_node_fused<<<kN / 16, 256, 0, stream>>>(h_in, n1h_gnw, n1h_gnb, n1h_msc,
                                                 mean_n, rstd_n, wt_abcd, bias_abcd, node_lin);
  k_edge_gate_fused<<<kE / 16, 192, 0, stream>>>(e_in, n1e_gnw, n1e_gnb, n1e_msc,
                                                 mean_ed, rstd_ed, wt_e, Eb, src, dst,
                                                 node_lin, agg_num, agg_den, e_out, gp);
  long nEl = (long)kN * kD;
  k_node_update<<<(int)((nEl + 255) / 256), 256, 0, stream>>>(h_in, node_lin, agg_num,
                                                              agg_den, h_out, gp);

  // GraphNorm 2 stats on updated h/e (living in d_out)
  k_stats_partial<<<sgrid, statsThreads, 0, stream>>>(h_out, kNPG, psum, psq);
  k_stats_final<<<finBlocks, 256, 0, stream>>>(kNPG, psum, psq, n2h_msc, mean_n, rstd_n);
  k_stats_partial<<<sgrid, statsThreads, 0, stream>>>(e_out, kEPG, psum, psq);
  k_stats_final<<<finBlocks, 256, 0, stream>>>(kEPG, psum, psq, n2e_msc, mean_ed, rstd_ed);

  // fused FFNs (norm fused, in-place residual accumulation into d_out)
  k_ffn_fused<<<kN / 16, 192, 0, stream>>>(h_out, kNPG, n2h_gnw, n2h_gnb, n2h_msc,
                                           mean_n, rstd_n, wt_f1h, f1hb, wt_f2h, f2hb, gp);
  k_ffn_fused<<<kE / 16, 192, 0, stream>>>(e_out, kEPG, n2e_gnw, n2e_gnb, n2e_msc,
                                           mean_ed, rstd_ed, wt_f1e, f1eb, wt_f2e, f2eb, gp);
}
